// MultiLayerCrossPatchBackbone_16209206575189
// MI455X (gfx1250) — matrix operand order repaired and checked by reading, not compiled, not run on hardware
//
#include <hip/hip_runtime.h>
#include <math.h>

// ---------------------------------------------------------------------------
// CDNA5 (gfx1250) wave32 WMMA types
// ---------------------------------------------------------------------------
typedef __attribute__((ext_vector_type(16))) __bf16 v16bf;
typedef __attribute__((ext_vector_type(8)))  float  v8f;

#define NEG_BIG (-1.0e30f)

static __device__ __forceinline__ unsigned short f2bf(float f) {
  union { float f; unsigned u; } x; x.f = f;
  unsigned r = x.u + 0x7FFFu + ((x.u >> 16) & 1u);   // round-to-nearest-even
  return (unsigned short)(r >> 16);
}

union FragB { v16bf v; unsigned u[8]; };

// 16-bit A-matrix 16x32 fragment (ISA 7.12.2): lane m = lane&15 is the row.
// grp=0: K = {0..7, 16..23}; grp=1: K = {8..15, 24..31}; VGPR j holds a K pair.
static __device__ __forceinline__ v16bf load_a_frag(const unsigned short* row, int grp) {
  FragB f;
#pragma unroll
  for (int j = 0; j < 8; ++j) {
    int kb = 2 * j + 8 * grp + ((j >= 4) ? 8 : 0);
    f.u[j] = *reinterpret_cast<const unsigned*>(row + kb);
  }
  return f.v;
}

// B-matrix 32x16 fragment where per-column data is K-contiguous in memory
// (lane n = column; grp=0: K = {0..7, 16..23}; grp=1: K = {8..15, 24..31}).
static __device__ __forceinline__ v16bf load_b_contig(const unsigned short* col, int grp) {
  FragB f;
#pragma unroll
  for (int j = 0; j < 8; ++j)
    f.u[j] = *reinterpret_cast<const unsigned*>(col + 2 * j + 8 * grp + ((j >= 4) ? 8 : 0));
  return f.v;
}

static __device__ __forceinline__ void store_out(float* p, float v) { *p = v; }
static __device__ __forceinline__ void store_out(unsigned short* p, float v) { *p = f2bf(v); }

// ---------------------------------------------------------------------------
// Kernel 1: 2D RoPE + fp32 -> bf16.  One thread per rotation pair.
// ---------------------------------------------------------------------------
__global__ void rope_bf16_kernel(const float* __restrict__ x,
                                 const int* __restrict__ coords,
                                 const int* __restrict__ image_size,
                                 const float* __restrict__ rope_cache,
                                 unsigned short* __restrict__ xr,
                                 int ntok) {
  int gid = blockIdx.x * blockDim.x + threadIdx.x;
  int tok = gid >> 7;          // 128 pairs per token (D=256)
  int p   = gid & 127;
  if (tok >= ntok) return;
  int img  = image_size[0];
  int axis = (p < 64) ? 1 : 0;           // x_part uses coords[...,1], y_part coords[...,0]
  int j    = (p < 64) ? p : (p - 64);
  int coord = coords[tok * 2 + axis];
  float cn = (float)coord / (float)img * 223.0f;   // (MAX_POS-1)=223
  cn = fminf(fmaxf(cn, 0.0f), 223.0f);
  int pos = (int)cn;
  float c = rope_cache[(pos * 64 + j) * 2 + 0];
  float s = rope_cache[(pos * 64 + j) * 2 + 1];
  int base = (p < 64) ? 0 : 128;
  size_t off = (size_t)tok * 256 + base + 2 * j;
  float p0 = x[off + 0];
  float p1 = x[off + 1];
  xr[off + 0] = f2bf(p0 * c - p1 * s);
  xr[off + 1] = f2bf(p0 * s + p1 * c);
}

// ---------------------------------------------------------------------------
// Kernel 2a: weight prep.  W[k][n] fp32 -> WT[n][k] bf16 (one-time, tiny).
// ---------------------------------------------------------------------------
__global__ void prep_wt_kernel(const float* __restrict__ W,
                               unsigned short* __restrict__ WT, int D) {
  int gid = blockIdx.x * blockDim.x + threadIdx.x;
  if (gid >= D * D) return;
  int n = gid / D, k = gid - n * D;
  WT[(size_t)n * D + k] = f2bf(W[(size_t)k * D + n]);
}

// ---------------------------------------------------------------------------
// Kernel 2b: V transpose.  Vrow[b*S+s][h*32+d] -> Vt[(b*8+h)*32+d][s]  (bf16)
// ---------------------------------------------------------------------------
__global__ void transpose_v_kernel(const unsigned short* __restrict__ Vrow,
                                   unsigned short* __restrict__ Vt, int S) {
  int gid = blockIdx.x * blockDim.x + threadIdx.x;   // over 8192*256
  int tok = gid >> 8, col = gid & 255;
  int b = tok / S, s = tok - b * S;
  int h = col >> 5, d = col & 31;
  Vt[((size_t)((b * 8 + h) * 32 + d)) * S + s] = Vrow[(size_t)tok * 256 + col];
}

// ---------------------------------------------------------------------------
// Kernel 3: C[M,N] = bf16 A[M,K] @ WT^T + bias.  WT is [N][K] bf16.
// Block = 4 waves; block computes a 64x64 tile; per K-chunk the 64x32 WT tile
// is staged into LDS with global_load_async_to_lds_b128 (ASYNCcnt path) and
// shared by all 4 waves.  Wave w -> rows [w*16, w*16+16), all 4 N-subtiles.
// ---------------------------------------------------------------------------
template <typename OUT>
__global__ void gemm_wmma_lds(const unsigned short* __restrict__ A,
                              const unsigned short* __restrict__ WT,
                              const float* __restrict__ bias,
                              OUT* __restrict__ C,
                              int M, int N, int K) {
  __shared__ __align__(16) unsigned short wtile[64 * 32];
  int nblocks_n = N >> 6;
  int mb = blockIdx.x / nblocks_n;
  int nb = blockIdx.x - mb * nblocks_n;
  int tid  = threadIdx.x;
  int wave = tid >> 5;
  int lane = tid & 31;
  int n = lane & 15, grp = lane >> 4;
  int row0 = mb * 64 + wave * 16;
  int col0 = nb * 64;

  v8f acc[4];
#pragma unroll
  for (int j = 0; j < 4; ++j) {
    float bv = bias[col0 + j * 16 + n];
#pragma unroll
    for (int r = 0; r < 8; ++r) acc[j][r] = bv;
  }

  const unsigned short* arow = A + (size_t)(row0 + n) * K;

  for (int kc = 0; kc < K; kc += 32) {
    __syncthreads();                       // previous tile fully consumed
    // async-stage WT[col0..col0+63][kc..kc+31] (4 KB) into LDS:
    // 256 x b128 transfers, 2 per thread.
#pragma unroll
    for (int it = 0; it < 2; ++it) {
      int id = tid + it * 128;
      int r  = id >> 2;                    // WT row (output column)
      int ch = id & 3;                     // 8-element chunk within the row
      unsigned ldsoff =
          (unsigned)(size_t)&wtile[r * 32 + ch * 8];
      unsigned long long gaddr =
          (unsigned long long)(size_t)(WT + (size_t)(col0 + r) * K + kc + ch * 8);
      asm volatile("global_load_async_to_lds_b128 %0, %1, off"
                   :: "v"(ldsoff), "v"(gaddr) : "memory");
    }
    asm volatile("s_wait_asynccnt 0" ::: "memory");
    __syncthreads();                       // LDS tile visible to all waves

    v16bf af = load_a_frag(arow + kc, grp);
#pragma unroll
    for (int j = 0; j < 4; ++j) {
      v16bf bf = load_b_contig(&wtile[(j * 16 + n) * 32], grp);
      acc[j] = __builtin_amdgcn_wmma_f32_16x16x32_bf16(false, af, false, bf,
                                                       (short)0, acc[j], false, false);
    }
  }

#pragma unroll
  for (int j = 0; j < 4; ++j)
#pragma unroll
    for (int r = 0; r < 8; ++r) {
      size_t row = (size_t)(row0 + r + 8 * grp);
      store_out(&C[row * N + col0 + j * 16 + n], acc[j][r]);
    }
}

// ---------------------------------------------------------------------------
// Kernel 4: flash attention, one wave per (b, h, 16-query tile), hd = 32.
// Q/K bf16 [B*S, 256] row-major (head h = cols h*32..h*32+31); V transposed
// as Vt[(b*8+h)*32+d][s].  mask: disallowed iff is_context[q] && !is_context[k].
// ---------------------------------------------------------------------------
__global__ void attn_fmha_kernel(const unsigned short* __restrict__ Q,
                                 const unsigned short* __restrict__ Km,
                                 const unsigned short* __restrict__ Vt,
                                 const unsigned char* __restrict__ ctx,
                                 unsigned short* __restrict__ O,
                                 int seq) {
  __shared__ __align__(16) unsigned short pl[16 * 32];   // P tile, bf16
  int qt = blockIdx.x;
  int bh = blockIdx.y;
  int b = bh >> 3, h = bh & 7;
  int lane = threadIdx.x & 31;
  int n = lane & 15, grp = lane >> 4;
  const float scale = 0.17677669529663687f;              // 1/sqrt(32)
  const size_t tokbase = (size_t)b * seq;
  const int hoff = h * 32;
  const unsigned short* vt0 = Vt + (size_t)(bh * 32 + n) * seq;        // d = n
  const unsigned short* vt1 = Vt + (size_t)(bh * 32 + 16 + n) * seq;   // d = 16+n

  // Q fragment: A-layout, K-dim = hd = 32 -> one fragment for the whole tile
  v16bf qf = load_a_frag(Q + (tokbase + qt * 16 + n) * 256 + hoff, grp);

  bool qc[8];
#pragma unroll
  for (int r = 0; r < 8; ++r)
    qc[r] = ctx[tokbase + qt * 16 + r + 8 * grp] != 0;

  float run_m[8], run_l[8];
  v8f acc0, acc1, zero;
#pragma unroll
  for (int r = 0; r < 8; ++r) {
    run_m[r] = NEG_BIG; run_l[r] = 0.f;
    acc0[r] = 0.f; acc1[r] = 0.f; zero[r] = 0.f;
  }

  for (int c0 = 0; c0 < seq; c0 += 32) {
    // S = Q K^T : key rows are d-contiguous -> B fragments straight from global
    v16bf kb0 = load_b_contig(Km + (tokbase + c0 + n) * 256 + hoff, grp);
    v16bf kb1 = load_b_contig(Km + (tokbase + c0 + 16 + n) * 256 + hoff, grp);
    v8f s0 = __builtin_amdgcn_wmma_f32_16x16x32_bf16(false, qf, false, kb0,
                                                     (short)0, zero, false, false);
    v8f s1 = __builtin_amdgcn_wmma_f32_16x16x32_bf16(false, qf, false, kb1,
                                                     (short)0, zero, false, false);
    bool kc0 = ctx[tokbase + c0 + n] != 0;
    bool kc1 = ctx[tokbase + c0 + 16 + n] != 0;

#pragma unroll
    for (int r = 0; r < 8; ++r) {
      bool m0 = qc[r] && !kc0;
      bool m1 = qc[r] && !kc1;
      float v0 = m0 ? NEG_BIG : s0[r] * scale;
      float v1 = m1 ? NEG_BIG : s1[r] * scale;
      // row-wise max across the 16-lane half (C-layout: row fixed per half)
      float t = fmaxf(v0, v1);
#pragma unroll
      for (int off = 1; off < 16; off <<= 1)
        t = fmaxf(t, __shfl_xor(t, off, 32));
      float nm   = fmaxf(run_m[r], t);
      float corr = __expf(run_m[r] - nm);
      float p0 = m0 ? 0.f : __expf(v0 - nm);
      float p1 = m1 ? 0.f : __expf(v1 - nm);
      float rs = p0 + p1;
#pragma unroll
      for (int off = 1; off < 16; off <<= 1)
        rs += __shfl_xor(rs, off, 32);
      run_l[r] = run_l[r] * corr + rs;
      run_m[r] = nm;
      acc0[r] *= corr;
      acc1[r] *= corr;
      pl[(r + 8 * grp) * 32 + n]      = f2bf(p0);
      pl[(r + 8 * grp) * 32 + 16 + n] = f2bf(p1);
    }
    asm volatile("s_wait_dscnt 0" ::: "memory");

    // re-read P as an A fragment (16 q x 32 keys, row-major stride 32 in LDS)
    v16bf pf = load_a_frag(pl + n * 32, grp);

    // V fragments from Vt: key-contiguous per output column -> dword loads
    v16bf vb0 = load_b_contig(vt0 + c0, grp);
    v16bf vb1 = load_b_contig(vt1 + c0, grp);
    acc0 = __builtin_amdgcn_wmma_f32_16x16x32_bf16(false, pf, false, vb0,
                                                   (short)0, acc0, false, false);
    acc1 = __builtin_amdgcn_wmma_f32_16x16x32_bf16(false, pf, false, vb1,
                                                   (short)0, acc1, false, false);
  }

#pragma unroll
  for (int r = 0; r < 8; ++r) {
    float inv = run_l[r] > 0.f ? 1.f / run_l[r] : 0.f;
    size_t row = tokbase + qt * 16 + r + 8 * grp;
    O[row * 256 + hoff + n]      = f2bf(acc0[r] * inv);
    O[row * 256 + hoff + 16 + n] = f2bf(acc1[r] * inv);
  }
}

// ---------------------------------------------------------------------------
// Launcher
// ---------------------------------------------------------------------------
extern "C" void kernel_launch(void* const* d_in, const int* in_sizes, int n_in,
                              void* d_out, int out_size, void* d_ws, size_t ws_size,
                              hipStream_t stream) {
  (void)in_sizes; (void)n_in; (void)out_size; (void)ws_size;
  const float*         x      = (const float*)d_in[0];
  const int*           coords = (const int*)d_in[1];
  const unsigned char* isctx  = (const unsigned char*)d_in[2];
  const int*           img    = (const int*)d_in[3];
  const float* Wq = (const float*)d_in[4];  const float* bq = (const float*)d_in[5];
  const float* Wk = (const float*)d_in[6];  const float* bk = (const float*)d_in[7];
  const float* Wv = (const float*)d_in[8];  const float* bv = (const float*)d_in[9];
  const float* Wo = (const float*)d_in[10]; const float* bo = (const float*)d_in[11];
  const float* rope = (const float*)d_in[12];
  float* out = (float*)d_out;

  const int B = 4, S = 2048, D = 256;
  const int M = B * S;                       // 8192 tokens

  char* ws = (char*)d_ws;                    // needs ~24.5 MB
  unsigned short* xr  = (unsigned short*)(ws);
  unsigned short* qb  = (unsigned short*)(ws + (((size_t)4)  << 20));
  unsigned short* kb  = (unsigned short*)(ws + (((size_t)8)  << 20));
  unsigned short* vb  = (unsigned short*)(ws + (((size_t)12) << 20));
  unsigned short* ao  = (unsigned short*)(ws + (((size_t)16) << 20));
  unsigned short* vt  = (unsigned short*)(ws + (((size_t)20) << 20));
  unsigned short* wtq = (unsigned short*)(ws + (((size_t)24) << 20));
  unsigned short* wtk = wtq + (size_t)D * D;
  unsigned short* wtv = wtk + (size_t)D * D;
  unsigned short* wto = wtv + (size_t)D * D;

  int npair = M * 128;
  rope_bf16_kernel<<<(npair + 255) / 256, 256, 0, stream>>>(x, coords, img, rope, xr, M);

  int wthreads = D * D;
  prep_wt_kernel<<<(wthreads + 255) / 256, 256, 0, stream>>>(Wq, wtq, D);
  prep_wt_kernel<<<(wthreads + 255) / 256, 256, 0, stream>>>(Wk, wtk, D);
  prep_wt_kernel<<<(wthreads + 255) / 256, 256, 0, stream>>>(Wv, wtv, D);
  prep_wt_kernel<<<(wthreads + 255) / 256, 256, 0, stream>>>(Wo, wto, D);

  int gblocks = (M / 64) * (D / 64);         // 512 blocks x 4 waves
  gemm_wmma_lds<unsigned short><<<gblocks, 128, 0, stream>>>(xr, wtq, bq, qb, M, D, D);
  gemm_wmma_lds<unsigned short><<<gblocks, 128, 0, stream>>>(xr, wtk, bk, kb, M, D, D);
  gemm_wmma_lds<unsigned short><<<gblocks, 128, 0, stream>>>(xr, wtv, bv, vb, M, D, D);

  int vthreads = M * D;
  transpose_v_kernel<<<(vthreads + 255) / 256, 256, 0, stream>>>(vb, vt, S);

  attn_fmha_kernel<<<dim3(S / 16, B * 8), 32, 0, stream>>>(qb, kb, vt, isctx, ao, S);

  gemm_wmma_lds<float><<<gblocks, 128, 0, stream>>>(ao, wto, bo, out, M, D, D);
}
